// NeuronMemory_70755291234743
// MI455X (gfx1250) — matrix operand order repaired and checked by reading, not compiled, not run on hardware
//
#include <hip/hip_runtime.h>
#include <stdint.h>

// Problem constants (from reference)
#define M_TOK 4096     // B*S
#define KDIM  1024     // D
#define NVOC  65536    // N
#define RDIM  128      // knowledge rank
#define CK    20       // coarse top-k
#define FK    10       // fine top-k

// Router GEMM tiling
#define BM 128
#define BN 128
#define BK 64
#define NSPLIT 16
#define NSPAN  (NVOC / NSPLIT)     // 4096 columns per workgroup
#define NPART  (NSPLIT * 2 * CK)   // 640 partial candidates per row

typedef __attribute__((ext_vector_type(16))) __bf16 v16bf;
typedef __attribute__((ext_vector_type(8)))  __bf16 v8bf;
typedef __attribute__((ext_vector_type(8)))  float  v8f;

__device__ inline __bf16 f2bf(float f) {
  union { float f; uint32_t u; } a; a.f = f;
  uint32_t r = a.u + 0x7FFFu + ((a.u >> 16) & 1u);   // round-to-nearest-even
  union { uint16_t s; __bf16 b; } o; o.s = (uint16_t)(r >> 16);
  return o.b;
}

template <int K>
__device__ inline void topk_insert(float (&ts)[K], int (&ti)[K], float v, int idx) {
  if (v <= ts[K - 1]) return;
  float cs = v; int ci = idx;
#pragma unroll
  for (int j = 0; j < K; ++j) {
    if (cs > ts[j]) {
      float a = ts[j]; int b = ti[j];
      ts[j] = cs; ti[j] = ci;
      cs = a; ci = b;
    }
  }
}

// ---------------------------------------------------------------------------
// CDNA5 async global->LDS copy path (guarded; sync fallback keeps compiles OK)
// Builtin signature (from probe diagnostic): pointers are int-vector typed,
// AS(1) source / AS(3) destination, then two int immediates (offset, cpol).
// ---------------------------------------------------------------------------
#if __has_builtin(__builtin_amdgcn_global_load_async_to_lds_b128)
#define HAVE_ASYNC_LDS 1
#endif

typedef int v4i_vs __attribute__((vector_size(16)));
typedef __attribute__((address_space(1))) v4i_vs* gvec4_p;
typedef __attribute__((address_space(3))) v4i_vs* lvec4_p;

__device__ inline void cp16(void* lds, const void* g) {
#ifdef HAVE_ASYNC_LDS
  __builtin_amdgcn_global_load_async_to_lds_b128(
      (gvec4_p)(uintptr_t)g,
      (lvec4_p)(uint32_t)(uintptr_t)lds,
      0, 0);
#else
  *(uint4*)lds = *(const uint4*)g;
#endif
}

__device__ inline void cp_wait() {
#ifdef HAVE_ASYNC_LDS
#if __has_builtin(__builtin_amdgcn_s_wait_asynccnt)
  __builtin_amdgcn_s_wait_asynccnt(0);
#else
  asm volatile("s_wait_asynccnt 0x0" ::: "memory");
#endif
#endif
}

// ---------------------------------------------------------------------------
// Pre-pass A: elementwise f32 -> bf16 (for x)
// ---------------------------------------------------------------------------
__global__ __launch_bounds__(256) void convert_bf16(
    const float* __restrict__ src, __bf16* __restrict__ dst) {
  size_t i = ((size_t)blockIdx.x * 256 + threadIdx.x) * 8;
  float4 a = *(const float4*)&src[i];
  float4 b = *(const float4*)&src[i + 4];
  union { __bf16 h[8]; uint4 u; } o;
  o.h[0] = f2bf(a.x); o.h[1] = f2bf(a.y); o.h[2] = f2bf(a.z); o.h[3] = f2bf(a.w);
  o.h[4] = f2bf(b.x); o.h[5] = f2bf(b.y); o.h[6] = f2bf(b.z); o.h[7] = f2bf(b.w);
  *(uint4*)&dst[i] = o.u;
}

// ---------------------------------------------------------------------------
// Pre-pass B: f32 [K][C] -> bf16 [C][K] (transpose+convert, 64x64 LDS tiles)
// ---------------------------------------------------------------------------
__global__ __launch_bounds__(256) void convert_transpose_bf16(
    const float* __restrict__ src, __bf16* __restrict__ dst,
    int kdim, int cols) {
  __shared__ __bf16 sT[64][64 + 8];   // 144B row stride (16B aligned)
  const int tid = threadIdx.x;
  const int k0 = blockIdx.y * 64, n0 = blockIdx.x * 64;
#pragma unroll
  for (int i = 0; i < 4; ++i) {
    int lin = tid + 256 * i;          // 0..1023 float4 chunks
    int kr = lin >> 4;                // 0..63
    int q  = lin & 15;                // 0..15
    float4 v = *(const float4*)&src[(size_t)(k0 + kr) * cols + n0 + q * 4];
    sT[q * 4 + 0][kr] = f2bf(v.x);
    sT[q * 4 + 1][kr] = f2bf(v.y);
    sT[q * 4 + 2][kr] = f2bf(v.z);
    sT[q * 4 + 3][kr] = f2bf(v.w);
  }
  __syncthreads();
  int nr = tid >> 2, q = tid & 3;     // 64 rows x 4 chunks of 16 halves
  uint4 a = *(const uint4*)&sT[nr][q * 16];
  uint4 b = *(const uint4*)&sT[nr][q * 16 + 8];
  uint4* o = (uint4*)&dst[(size_t)(n0 + nr) * kdim + k0 + q * 16];
  o[0] = a; o[1] = b;
}

// ---------------------------------------------------------------------------
// Tile fill: 128 rows x 128 bytes each for A and B (16B chunks, 4+4/thread)
// ---------------------------------------------------------------------------
__device__ inline void load_tiles(const __bf16* __restrict__ A,
                                  const __bf16* __restrict__ Bt,
                                  __bf16 (*sAp)[BK + 8], __bf16 (*sBp)[BK + 8],
                                  int m0, int n0, int kk, int tid) {
#pragma unroll
  for (int i = 0; i < 4; ++i) {
    int id = tid + 256 * i;           // 0..1023
    int r = id >> 3;                  // 0..127
    int q = id & 7;                   // 8 x 16B = 128B = 64 halves
    cp16(&sAp[r][q * 8], A + (size_t)(m0 + r) * KDIM + kk + q * 8);
    cp16(&sBp[r][q * 8], Bt + (size_t)(n0 + r) * KDIM + kk + q * 8);
  }
}

// ---------------------------------------------------------------------------
// Kernel 1: router GEMM (bf16 WMMA, f32 acc), double-buffered async tiles,
// fused running per-row top-20. Logits never touch HBM.
// ---------------------------------------------------------------------------
__global__ __launch_bounds__(256) void router_gemm_topk(
    const __bf16* __restrict__ Abf, const __bf16* __restrict__ Wt,
    float* __restrict__ pScore, int* __restrict__ pIdx) {
  __shared__ __bf16 sA[2][BM][BK + 8];
  __shared__ __bf16 sB[2][BN][BK + 8];
  __shared__ float  sC[BM][BN + 1];

  const int tid  = threadIdx.x;
  const int wave = tid >> 5;
  const int lane = tid & 31;
  const int m0   = blockIdx.x * BM;       // fast dim: co-resident WGs share Wt slice
  const int nsplit = blockIdx.y;
  const int nbase  = nsplit * NSPAN;

  const int row  = tid & (BM - 1);
  const int half = tid >> 7;

  const int wm = wave & 3;                // rows wm*32 .. +32
  const int wn = wave >> 2;               // cols wn*64 .. +64
  const int kh = lane >> 4;
  const int l15 = lane & 15;

  float ts[CK]; int ti[CK];
#pragma unroll
  for (int j = 0; j < CK; ++j) { ts[j] = -INFINITY; ti[j] = 0; }

  for (int n0 = nbase; n0 < nbase + NSPAN; n0 += BN) {
    v8f c[2][4];
#pragma unroll
    for (int s = 0; s < 2; ++s)
#pragma unroll
      for (int t = 0; t < 4; ++t)
#pragma unroll
        for (int i = 0; i < 8; ++i) c[s][t][i] = 0.0f;

    load_tiles(Abf, Wt, sA[0], sB[0], m0, n0, 0, tid);
    int buf = 0;
    for (int kk = 0; kk < KDIM; kk += BK, buf ^= 1) {
      cp_wait();
      __syncthreads();
      if (kk + BK < KDIM)
        load_tiles(Abf, Wt, sA[buf ^ 1], sB[buf ^ 1], m0, n0, kk + BK, tid);

#pragma unroll
      for (int ks = 0; ks < 2; ++ks) {    // two 16x16x32 substeps of BK=64
        const int kb = ks * 32;
#pragma unroll
        for (int s = 0; s < 2; ++s) {
          int arow = wm * 32 + s * 16 + l15;
          v8bf alo = *(const v8bf*)&sA[buf][arow][kb + kh * 16];
          v8bf ahi = *(const v8bf*)&sA[buf][arow][kb + kh * 16 + 8];
          v16bf af = __builtin_shufflevector(alo, ahi, 0, 1, 2, 3, 4, 5, 6, 7,
                                             8, 9, 10, 11, 12, 13, 14, 15);
#pragma unroll
          for (int t = 0; t < 4; ++t) {
            int bcol = wn * 64 + t * 16 + l15;
            v8bf blo = *(const v8bf*)&sB[buf][bcol][kb + kh * 16];
            v8bf bhi = *(const v8bf*)&sB[buf][bcol][kb + kh * 16 + 8];
            v16bf bfr = __builtin_shufflevector(blo, bhi, 0, 1, 2, 3, 4, 5, 6, 7,
                                                8, 9, 10, 11, 12, 13, 14, 15);
            c[s][t] = __builtin_amdgcn_wmma_f32_16x16x32_bf16(
                false, af, false, bfr, (short)0, c[s][t], false, false);
          }
        }
      }
    }
    __syncthreads();

    // stage C tile: VGPR i of lane L -> (M = i + 8*(L>>4), N = L&15)
#pragma unroll
    for (int s = 0; s < 2; ++s)
#pragma unroll
      for (int t = 0; t < 4; ++t)
#pragma unroll
        for (int i = 0; i < 8; ++i)
          sC[wm * 32 + s * 16 + 8 * kh + i][wn * 64 + t * 16 + l15] = c[s][t][i];
    __syncthreads();

    // running top-20: 2 threads/row, 64 cols each
    for (int j = 0; j < 64; ++j) {
      int col = half * 64 + j;
      topk_insert<CK>(ts, ti, sC[row][col], n0 + col);
    }
    __syncthreads();
  }

  size_t base = (size_t)(m0 + row) * NPART + (size_t)(nsplit * 2 + half) * CK;
#pragma unroll
  for (int j = 0; j < CK; ++j) {
    pScore[base + j] = ts[j];
    pIdx[base + j]   = ti[j];
  }
}

// ---------------------------------------------------------------------------
// Kernel 2: merge partial top-20 lists -> final coarse-20 indices per row
// ---------------------------------------------------------------------------
__global__ __launch_bounds__(256) void merge_topk(
    const float* __restrict__ pScore, const int* __restrict__ pIdx,
    int* __restrict__ cIdx) {
  int rowg = blockIdx.x * 256 + threadIdx.x;
  float ts[CK]; int ti[CK];
#pragma unroll
  for (int j = 0; j < CK; ++j) { ts[j] = -INFINITY; ti[j] = 0; }
  const float* ps = pScore + (size_t)rowg * NPART;
  const int*   pi = pIdx + (size_t)rowg * NPART;
  for (int j = 0; j < NPART; ++j)
    topk_insert<CK>(ts, ti, ps[j], pi[j]);
#pragma unroll
  for (int j = 0; j < CK; ++j) cIdx[(size_t)rowg * CK + j] = ti[j];
}

// ---------------------------------------------------------------------------
// Kernel 3: query = x @ W_enc (bf16 WMMA, Wet pre-transposed [128][1024])
// ---------------------------------------------------------------------------
__global__ __launch_bounds__(256) void query_gemm(
    const __bf16* __restrict__ Abf, const __bf16* __restrict__ Wet,
    float* __restrict__ q) {
  __shared__ __bf16 sA[2][BM][BK + 8];
  __shared__ __bf16 sB[2][RDIM][BK + 8];

  const int tid  = threadIdx.x;
  const int wave = tid >> 5;
  const int lane = tid & 31;
  const int m0   = blockIdx.x * BM;
  const int wm = wave & 3, wn = wave >> 2;
  const int kh = lane >> 4, l15 = lane & 15;

  v8f c[2][4];
#pragma unroll
  for (int s = 0; s < 2; ++s)
#pragma unroll
    for (int t = 0; t < 4; ++t)
#pragma unroll
      for (int i = 0; i < 8; ++i) c[s][t][i] = 0.0f;

  load_tiles(Abf, Wet, sA[0], sB[0], m0, 0, 0, tid);
  int buf = 0;
  for (int kk = 0; kk < KDIM; kk += BK, buf ^= 1) {
    cp_wait();
    __syncthreads();
    if (kk + BK < KDIM)
      load_tiles(Abf, Wet, sA[buf ^ 1], sB[buf ^ 1], m0, 0, kk + BK, tid);

#pragma unroll
    for (int ks = 0; ks < 2; ++ks) {
      const int kb = ks * 32;
#pragma unroll
      for (int s = 0; s < 2; ++s) {
        int arow = wm * 32 + s * 16 + l15;
        v8bf alo = *(const v8bf*)&sA[buf][arow][kb + kh * 16];
        v8bf ahi = *(const v8bf*)&sA[buf][arow][kb + kh * 16 + 8];
        v16bf af = __builtin_shufflevector(alo, ahi, 0, 1, 2, 3, 4, 5, 6, 7,
                                           8, 9, 10, 11, 12, 13, 14, 15);
#pragma unroll
        for (int t = 0; t < 4; ++t) {
          int bcol = wn * 64 + t * 16 + l15;
          v8bf blo = *(const v8bf*)&sB[buf][bcol][kb + kh * 16];
          v8bf bhi = *(const v8bf*)&sB[buf][bcol][kb + kh * 16 + 8];
          v16bf bfr = __builtin_shufflevector(blo, bhi, 0, 1, 2, 3, 4, 5, 6, 7,
                                              8, 9, 10, 11, 12, 13, 14, 15);
          c[s][t] = __builtin_amdgcn_wmma_f32_16x16x32_bf16(
              false, af, false, bfr, (short)0, c[s][t], false, false);
        }
      }
    }
  }

#pragma unroll
  for (int s = 0; s < 2; ++s)
#pragma unroll
    for (int t = 0; t < 4; ++t)
#pragma unroll
      for (int i = 0; i < 8; ++i)
        q[(size_t)(m0 + wm * 32 + s * 16 + 8 * kh + i) * RDIM +
          wn * 64 + t * 16 + l15] = c[s][t][i];
}

// ---------------------------------------------------------------------------
// Kernel 4: per token — 20 query.K dots, top-10, softmax, weighted V gather
// ---------------------------------------------------------------------------
__global__ __launch_bounds__(256) void fine_kernel(
    const float* __restrict__ q, const int* __restrict__ cIdx,
    const float* __restrict__ K_all, const float* __restrict__ V_all,
    float* __restrict__ out) {
  const int wave = threadIdx.x >> 5;
  const int lane = threadIdx.x & 31;
  const int m = blockIdx.x * 8 + wave;

  float q0[4];
#pragma unroll
  for (int j = 0; j < 4; ++j) q0[j] = q[(size_t)m * RDIM + lane * 4 + j];

  float sc[CK]; int cid[CK];
#pragma unroll
  for (int c = 0; c < CK; ++c) {
    int kr = cIdx[(size_t)m * CK + c];
    cid[c] = kr;
    const float* krow = &K_all[(size_t)kr * RDIM + lane * 4];
    float s = q0[0] * krow[0] + q0[1] * krow[1] + q0[2] * krow[2] + q0[3] * krow[3];
#pragma unroll
    for (int off = 16; off >= 1; off >>= 1) s += __shfl_xor(s, off, 32);
    sc[c] = s * 0.08838834764831845f;   // 1/sqrt(128)
  }

  float ts[FK]; int ti[FK];
#pragma unroll
  for (int j = 0; j < FK; ++j) { ts[j] = -INFINITY; ti[j] = 0; }
#pragma unroll
  for (int c = 0; c < CK; ++c) topk_insert<FK>(ts, ti, sc[c], cid[c]);

  float w[FK], den = 0.0f;
#pragma unroll
  for (int f = 0; f < FK; ++f) { w[f] = __expf(ts[f] - ts[0]); den += w[f]; }
  float inv = 1.0f / den;

  float acc[32];
#pragma unroll
  for (int k = 0; k < 32; ++k) acc[k] = 0.0f;
#pragma unroll
  for (int f = 0; f < FK; ++f) {
    const float* vr = &V_all[(size_t)ti[f] * KDIM];
    float wf = w[f] * inv;
#pragma unroll
    for (int k = 0; k < 32; ++k) acc[k] += wf * vr[lane + 32 * k];
  }
#pragma unroll
  for (int k = 0; k < 32; ++k) out[(size_t)m * KDIM + lane + 32 * k] = acc[k];
}

// ---------------------------------------------------------------------------
extern "C" void kernel_launch(void* const* d_in, const int* in_sizes, int n_in,
                              void* d_out, int out_size, void* d_ws, size_t ws_size,
                              hipStream_t stream) {
  const float* x     = (const float*)d_in[0];
  const float* Wr    = (const float*)d_in[1];
  const float* We    = (const float*)d_in[2];
  const float* K_all = (const float*)d_in[3];
  const float* V_all = (const float*)d_in[4];
  float* out = (float*)d_out;

  char* ws = (char*)d_ws;
  size_t off = 0;
  __bf16* Abf = (__bf16*)(ws + off); off += (size_t)M_TOK * KDIM * 2;   // 8 MB
  __bf16* Wt  = (__bf16*)(ws + off); off += (size_t)NVOC * KDIM * 2;    // 128 MB
  __bf16* Wet = (__bf16*)(ws + off); off += (size_t)RDIM * KDIM * 2;    // 256 KB
  float* pScore = (float*)(ws + off); off += (size_t)M_TOK * NPART * 4; // 10.5 MB
  int*   pIdx   = (int*)(ws + off);   off += (size_t)M_TOK * NPART * 4; // 10.5 MB
  int*   cIdx   = (int*)(ws + off);   off += (size_t)M_TOK * CK * 4;    // 320 KB
  float* query  = (float*)(ws + off);                                   // 2 MB

  convert_bf16<<<(M_TOK * KDIM) / (256 * 8), 256, 0, stream>>>(x, Abf);
  convert_transpose_bf16<<<dim3(NVOC / 64, KDIM / 64), 256, 0, stream>>>(
      Wr, Wt, KDIM, NVOC);
  convert_transpose_bf16<<<dim3(RDIM / 64, KDIM / 64), 256, 0, stream>>>(
      We, Wet, KDIM, RDIM);

  router_gemm_topk<<<dim3(M_TOK / BM, NSPLIT), 256, 0, stream>>>(
      Abf, Wt, pScore, pIdx);
  merge_topk<<<M_TOK / 256, 256, 0, stream>>>(pScore, pIdx, cIdx);
  query_gemm<<<M_TOK / BM, 256, 0, stream>>>(Abf, Wet, query);
  fine_kernel<<<M_TOK / 8, 256, 0, stream>>>(query, cIdx, K_all, V_all, out);
}
